// GraphAttentionLayer_89481348644986
// MI455X (gfx1250) — matrix operand order repaired and checked by reading, not compiled, not run on hardware
//
#include <hip/hip_runtime.h>

#define NNODES 50000
#define NEDGES 800000
#define NDIM   128
#define EDIM   64
#define MDIM   128
#define INDIM  320
#define LN_EPS 1e-5f

typedef __attribute__((ext_vector_type(16))) __bf16       v16bf;
typedef __attribute__((ext_vector_type(8)))  float        v8f;
typedef __attribute__((ext_vector_type(8)))  unsigned int v8u;

// pack two f32 -> packed bf16 {f_lo in [15:0], f_hi in [31:16]}, round-half-up:
// one add per source + a single v_perm_b32 grabbing both high halves.
__device__ __forceinline__ unsigned int pack_bf16(float f_lo, float f_hi) {
  unsigned int ul = __float_as_uint(f_lo) + 0x8000u;
  unsigned int uh = __float_as_uint(f_hi) + 0x8000u;
  return __builtin_amdgcn_perm(uh, ul, 0x07060302u);  // {uh[31:16], ul[31:16]}
}

__device__ __forceinline__ unsigned short f2bf(float f) {
  unsigned int u = __float_as_uint(f);
  u += 0x7fffu + ((u >> 16) & 1u);   // RNE (used only in tiny weight-prep kernel)
  return (unsigned short)(u >> 16);
}

// x[e][K] = concat(nodes[snd], nodes[rcv], edges[e]); K multiple of 8 stays in one region
__device__ __forceinline__ const float* xsrc(const float* __restrict__ nodes,
                                             const float* __restrict__ edges,
                                             int snd, int rcv, long e, int K) {
  if (K < NDIM)     return nodes + (long)snd * NDIM + K;
  if (K < 2 * NDIM) return nodes + (long)rcv * NDIM + (K - NDIM);
  return edges + e * EDIM + (K - 2 * NDIM);
}

// Build bf16 A fragments (16x32 per K-step, 10 steps covering K=320).
// A layout used here (contiguous k): lane holds row M=lane&15; lanes<16: K={b..b+15} with b=32*ks;
// lanes>=16: same shifted by +16.
__device__ __forceinline__ void gather_a(const float* __restrict__ nodes,
                                         const int* __restrict__ eidx,
                                         const float* __restrict__ edges,
                                         long e0, int hi, int col, v16bf a[10]) {
  long e = e0 + col;
  int snd = eidx[e];
  int rcv = eidx[(long)NEDGES + e];
#pragma unroll
  for (int ks = 0; ks < 10; ++ks) {
    unsigned int pk[8];
#pragma unroll
    for (int g = 0; g < 2; ++g) {
      int K = 32 * ks + 16 * hi + 8 * g;
      const float4* p = (const float4*)xsrc(nodes, edges, snd, rcv, e, K);
      float4 u0 = p[0];
      float4 u1 = p[1];
      pk[g * 4 + 0] = pack_bf16(u0.x, u0.y);
      pk[g * 4 + 1] = pack_bf16(u0.z, u0.w);
      pk[g * 4 + 2] = pack_bf16(u1.x, u1.y);
      pk[g * 4 + 3] = pack_bf16(u1.z, u1.w);
    }
    v8u pv = { pk[0], pk[1], pk[2], pk[3], pk[4], pk[5], pk[6], pk[7] };
    a[ks] = __builtin_bit_cast(v16bf, pv);
  }
}

// GEMM: D(16x128) = A(16x320) * W(320x128), W pre-transposed bf16 [N][K].
// K-step outer / N-tile inner: 8 consecutive WMMAs share one A fragment.
__device__ __forceinline__ void wmma_gemm(const v16bf a[10],
                                          const unsigned short* __restrict__ wt,
                                          int hi, int col, v8f acc[8]) {
#pragma unroll
  for (int t = 0; t < 8; ++t) acc[t] = (v8f){0.f,0.f,0.f,0.f,0.f,0.f,0.f,0.f};
#pragma unroll
  for (int ks = 0; ks < 10; ++ks) {
#pragma unroll
    for (int t = 0; t < 8; ++t) {
      // dense B 32x16 bf16 layout: lane N=col, K = 32*ks + 16*hi + j (contiguous 16)
      const v8u* bp = (const v8u*)(wt + ((t * 16 + col) * INDIM + ks * 32 + hi * 16));
      v16bf b = __builtin_bit_cast(v16bf, *bp);
      acc[t] = __builtin_amdgcn_wmma_f32_16x16x32_bf16(false, a[ks], false, b,
                                                       (short)0, acc[t], false, false);
    }
  }
}

// ---------------- Pass A: h = LN(relu(x@W1+b1)); logits = relu(h@W2+b2); seg max ----
__global__ __launch_bounds__(128) void k_logits(
    const float* __restrict__ nodes, const int* __restrict__ eidx,
    const float* __restrict__ edges, const unsigned short* __restrict__ w1t,
    const float* __restrict__ b1, const float* __restrict__ gamma,
    const float* __restrict__ beta, const float* __restrict__ W2,
    const float* __restrict__ b2, float* __restrict__ logits,
    unsigned int* __restrict__ smax)
{
  int gtid = blockIdx.x * blockDim.x + threadIdx.x;
  int wave = gtid >> 5;
  int lane = threadIdx.x & 31;
  if (wave >= NEDGES / 16) return;
  int hi = lane >> 4, col = lane & 15;
  long e0 = (long)wave * 16;

  v16bf a[10];
  gather_a(nodes, eidx, edges, e0, hi, col, a);

  v8f h[8];
  wmma_gemm(a, w1t, hi, col, h);

  // bias + relu (C/D layout: h[t][r] = edge M=r+8*hi, channel N=col+16*t)
#pragma unroll
  for (int t = 0; t < 8; ++t) {
    float bn = b1[col + 16 * t];
#pragma unroll
    for (int r = 0; r < 8; ++r) {
      float v = h[t][r] + bn;
      h[t][r] = v > 0.f ? v : 0.f;
    }
  }
  // LayerNorm stats: reduce over 128 channels = 8 regs x 16 lanes of a half-group
  float mu[8], rs[8];
#pragma unroll
  for (int r = 0; r < 8; ++r) {
    float s = 0.f, s2 = 0.f;
#pragma unroll
    for (int t = 0; t < 8; ++t) { float v = h[t][r]; s += v; s2 += v * v; }
#pragma unroll
    for (int m = 1; m <= 8; m <<= 1) {
      s  += __shfl_xor(s,  m, 32);
      s2 += __shfl_xor(s2, m, 32);
    }
    float mean = s * (1.f / 128.f);
    mu[r] = mean;
    rs[r] = rsqrtf(s2 * (1.f / 128.f) - mean * mean + LN_EPS);
  }
  float gv[8], bv[8], wv[8];
#pragma unroll
  for (int t = 0; t < 8; ++t) {
    int N = col + 16 * t;
    gv[t] = gamma[N]; bv[t] = beta[N]; wv[t] = W2[N];
  }
  float bias2 = b2[0];
  float lg[8];
#pragma unroll
  for (int r = 0; r < 8; ++r) {
    float d = 0.f;
#pragma unroll
    for (int t = 0; t < 8; ++t) {
      float v = (h[t][r] - mu[r]) * rs[r] * gv[t] + bv[t];
      d += v * wv[t];
    }
#pragma unroll
    for (int m = 1; m <= 8; m <<= 1) d += __shfl_xor(d, m, 32);
    d += bias2;
    lg[r] = d > 0.f ? d : 0.f;
  }
  if (col < 8) {
    float v = 0.f;
#pragma unroll
    for (int r = 0; r < 8; ++r) v = (col == r) ? lg[r] : v;
    long ee = e0 + 8 * hi + col;
    logits[ee] = v;
    int rr = eidx[(long)NEDGES + ee];
    // logits >= 0, so uint-bit max == float max; smax init 0 matches isfinite->0 semantics
    atomicMax(&smax[rr], __float_as_uint(v));
  }
}

// ---------------- Pass B: exp and segment sum ----------------
__global__ void k_expsum(const float* __restrict__ logits, const int* __restrict__ eidx,
                         const unsigned int* __restrict__ smax,
                         float* __restrict__ expv, float* __restrict__ ssum)
{
  long e = (long)blockIdx.x * blockDim.x + threadIdx.x;
  if (e >= NEDGES) return;
  int rr = eidx[(long)NEDGES + e];
  float m = __uint_as_float(smax[rr]);
  float x = expf(logits[e] - m);
  expv[e] = x;
  atomicAdd(&ssum[rr], x);
}

// ---------------- Pass C: queries = x@Wq + bq, weighted atomic scatter ----------------
__global__ __launch_bounds__(128) void k_scatter(
    const float* __restrict__ nodes, const int* __restrict__ eidx,
    const float* __restrict__ edges, const unsigned short* __restrict__ wqt,
    const float* __restrict__ bq, const float* __restrict__ expv,
    const float* __restrict__ ssum, float* __restrict__ out)
{
  int gtid = blockIdx.x * blockDim.x + threadIdx.x;
  int wave = gtid >> 5;
  int lane = threadIdx.x & 31;
  if (wave >= NEDGES / 16) return;
  int hi = lane >> 4, col = lane & 15;
  long e0 = (long)wave * 16;

  v16bf a[10];
  gather_a(nodes, eidx, edges, e0, hi, col, a);

  v8f q[8];
  wmma_gemm(a, wqt, hi, col, q);

  float wgt[8];
  int rcvr[8];
#pragma unroll
  for (int r = 0; r < 8; ++r) {
    long ee = e0 + 8 * hi + r;   // this half-group's 8 edges
    int rr = eidx[(long)NEDGES + ee];
    rcvr[r] = rr;
    wgt[r] = expv[ee] / (ssum[rr] + 1e-10f);
  }
#pragma unroll
  for (int t = 0; t < 8; ++t) {
    int N = col + 16 * t;
    float bn = bq[N];
#pragma unroll
    for (int r = 0; r < 8; ++r) {
      float val = (q[t][r] + bn) * wgt[r];
      atomicAdd(out + (long)rcvr[r] * MDIM + N, val);
    }
  }
}

// ---------------- helpers ----------------
__global__ void k_leaky(float* __restrict__ out, long n) {
  long i = (long)blockIdx.x * blockDim.x + threadIdx.x;
  if (i < n) { float v = out[i]; out[i] = v > 0.f ? v : 0.01f * v; }
}

// vectorized zero (n must be a multiple of 4 dwords)
__global__ void k_zero4(uint4* __restrict__ p, long n4) {
  long i = (long)blockIdx.x * blockDim.x + threadIdx.x;
  if (i < n4) p[i] = (uint4){0u, 0u, 0u, 0u};
}

// transpose + bf16-convert both weight matrices: wt[n*320 + k] = bf16(W[k*128 + n])
__global__ void k_prepw(const float* __restrict__ Wq, const float* __restrict__ W1,
                        unsigned short* __restrict__ wqt, unsigned short* __restrict__ w1t) {
  int i = blockIdx.x * blockDim.x + threadIdx.x;
  if (i >= INDIM * MDIM) return;
  int n = i / INDIM, k = i % INDIM;
  wqt[i] = f2bf(Wq[k * MDIM + n]);
  w1t[i] = f2bf(W1[k * MDIM + n]);
}

extern "C" void kernel_launch(void* const* d_in, const int* in_sizes, int n_in,
                              void* d_out, int out_size, void* d_ws, size_t ws_size,
                              hipStream_t stream) {
  const float* nodes = (const float*)d_in[0];
  const int*   eidx  = (const int*)d_in[1];
  const float* edges = (const float*)d_in[2];
  const float* Wq    = (const float*)d_in[3];
  const float* bq    = (const float*)d_in[4];
  const float* W1    = (const float*)d_in[5];
  const float* b1    = (const float*)d_in[6];
  const float* gamma = (const float*)d_in[7];
  const float* beta  = (const float*)d_in[8];
  const float* W2    = (const float*)d_in[9];
  const float* b2    = (const float*)d_in[10];
  float* out = (float*)d_out;

  // workspace layout (~7 MB total)
  char* ws = (char*)d_ws;
  unsigned short* wqt  = (unsigned short*)(ws);                         // 81920 B
  unsigned short* w1t  = (unsigned short*)(ws + 81920);                 // 81920 B
  float*          logits = (float*)(ws + 163840);                       // 3.2 MB
  float*          expv   = (float*)(ws + 163840 + 3200000);             // 3.2 MB
  unsigned int*   smax   = (unsigned int*)(ws + 163840 + 6400000);      // 200 KB
  float*          ssum   = (float*)(ws + 163840 + 6400000 + 200000);    // 200 KB

  const long outN = (long)NNODES * MDIM;  // 6.4M floats

  // fresh state every call (graph-replay safe): zero output acc + seg buffers
  k_zero4<<<(int)((outN / 4 + 255) / 256), 256, 0, stream>>>((uint4*)out, outN / 4);
  k_zero4<<<(2 * NNODES / 4 + 255) / 256, 256, 0, stream>>>(
      (uint4*)smax, 2 * NNODES / 4);  // smax+ssum contiguous, 400000 B
  k_prepw<<<(INDIM * MDIM + 255) / 256, 256, 0, stream>>>(Wq, W1, wqt, w1t);

  int ntiles = NEDGES / 16;   // 50000 edge tiles, one wave each
  int blocks = ntiles / 4;    // 4 waves (128 threads) per block

  k_logits<<<blocks, 128, 0, stream>>>(nodes, eidx, edges, w1t, b1, gamma, beta,
                                       W2, b2, logits, smax);
  k_expsum<<<(NEDGES + 255) / 256, 256, 0, stream>>>(logits, eidx, smax, expv, ssum);
  k_scatter<<<blocks, 128, 0, stream>>>(nodes, eidx, edges, wqt, bq, expv, ssum, out);
  k_leaky<<<(int)((outN + 255) / 256), 256, 0, stream>>>(out, outN);
}
